// ConvSSMPointTrackingHead_13091060318340
// MI455X (gfx1250) — matrix operand order repaired and checked by reading, not compiled, not run on hardware
//
#include <hip/hip_runtime.h>
#include <hip/hip_bf16.h>
#include <math.h>

// ---------------------------------------------------------------------------
// ConvSSMPointTrackingHead for MI455X (gfx1250, wave32, WMMA).
//
//   x = video @ conv_w + conv_b                        (WMMA bf16 GEMM)
//   3x blocks:
//     R(c,f)   = (sum_{k=0..7} A_f^k) * B_f            (direct tap DFT, Horner)
//     X_f      = 3D DFT(x)  (separable 32/32/8-pt)
//     h        = Re(IDFT(X_f * R))
//     x       += LayerNorm(h)                          (LDS reduction)
//     m        = gelu(x @ W1 + b1)                     (WMMA, bf16 out fused)
//     x       += m @ W2 + b2                           (WMMA, residual fused)
//   feat = mean_{H,W}(x);  heads -> trajectories, occlusion
//
// GEMM: block = 8 waves = 64x128 macro-tile; wave = 32x32 (2x2 WMMA tiles).
// B panel (8KB/K-step) staged in LDS with GLOBAL_LOAD_ASYNC_TO_LDS_B128
// (ASYNCcnt) and consumed via ds_load_b128 by all 8 waves.
// ---------------------------------------------------------------------------

typedef __attribute__((ext_vector_type(16))) __bf16 v16bf;
typedef __attribute__((ext_vector_type(8)))  float  v8f;

union BV16 { v16bf v; uint4 q[2]; };

__device__ __forceinline__ unsigned short f2bf(float f) {
  union { float f; unsigned int u; } x; x.f = f;
  unsigned int u = x.u + 0x7FFFu + ((x.u >> 16) & 1u);   // round-to-nearest-even
  return (unsigned short)(u >> 16);
}

// ------------------------------ conversions --------------------------------

__global__ void f2bf_kernel(const float* __restrict__ in,
                            unsigned short* __restrict__ out, int n) {
  int id = blockIdx.x * blockDim.x + threadIdx.x;
  if (id < n) out[id] = f2bf(in[id]);
}

// Pack W[K][N] (f32 row-major) into Pb[kt][n][32] bf16 so each lane reads 32
// contiguous bytes: element e in [0,32) holds K = kt*32 + e (lane half selects
// e<16 vs e>=16, the same contiguous K split {0..15}/{16..31} as the A loads).
__global__ void pack_w_kernel(const float* __restrict__ W,
                              unsigned short* __restrict__ Pb, int K, int N) {
  int idx = blockIdx.x * blockDim.x + threadIdx.x;
  int total = (K >> 5) * N * 32;
  if (idx >= total) return;
  int e  = idx & 31;
  int n  = (idx >> 5) % N;
  int kt = idx / (32 * N);
  int k  = kt * 32 + e;
  Pb[idx] = f2bf(W[(size_t)k * N + n]);
}

// ------------------------------ WMMA GEMM ----------------------------------
// EPI: 0 = +bias -> f32 ; 1 = +bias, gelu -> bf16 ; 2 = +bias +resid -> f32
// Grid: (M/64, N/128), block 256 threads (8 waves). Wave (wm,wn) computes the
// 32x32 sub-tile; B panel for the current K-step lives in LDS.
template <int EPI>
__global__ __launch_bounds__(256) void gemm_wmma_kernel(
    const unsigned short* __restrict__ A, const unsigned short* __restrict__ Bp,
    const float* __restrict__ bias, const float* __restrict__ resid,
    float* __restrict__ outF, unsigned short* __restrict__ outH,
    int M, int N, int K) {
  __shared__ uint4 panel[512];                       // 8KB: 128 cols x 64B
  int tid  = threadIdx.x;
  int lane = tid & 31;
  int wave = tid >> 5;
  int wm = wave >> 2, wn = wave & 3;                 // 2 x 4 wave grid
  int l  = lane & 15;
  int hs = (lane >> 4) & 1;                          // lane half -> K sub-block
  int bm = blockIdx.x, bn = blockIdx.y;

  int row0 = bm * 64 + wm * 32 + l;
  const uint4* Arow0 = (const uint4*)(A + (size_t)row0 * K);        // 8 bf16/u4
  const uint4* Arow1 = (const uint4*)(A + (size_t)(row0 + 16) * K);

  unsigned int lds0 =
      (unsigned int)(unsigned long long)(void*)panel + tid * 16;
  const unsigned char* BpB = (const unsigned char*)Bp;

  v8f acc00 = {0.f,0.f,0.f,0.f,0.f,0.f,0.f,0.f};
  v8f acc01 = {0.f,0.f,0.f,0.f,0.f,0.f,0.f,0.f};
  v8f acc10 = {0.f,0.f,0.f,0.f,0.f,0.f,0.f,0.f};
  v8f acc11 = {0.f,0.f,0.f,0.f,0.f,0.f,0.f,0.f};

  int ktiles = K >> 5;
  for (int kt = 0; kt < ktiles; ++kt) {
    // Async-stage the 8KB B panel: 256 threads x 2 x b128. INST_OFFSET is
    // added to BOTH the LDS and global addresses (ISA 10.x), so chunk 2 is
    // just offset:4096 on the same address registers.
    unsigned long long g0 = (unsigned long long)(
        BpB + (size_t)64 * ((size_t)kt * N + (size_t)bn * 128) + tid * 16);
    asm volatile("global_load_async_to_lds_b128 %0, %1, off"
                 :: "v"(lds0), "v"(g0) : "memory");
    asm volatile("global_load_async_to_lds_b128 %0, %1, off offset:4096"
                 :: "v"(lds0), "v"(g0) : "memory");
    asm volatile("s_wait_asynccnt 0" ::: "memory");
    __syncthreads();                                 // panel valid for block

    BV16 a0, a1, b0, b1;
    int ai = kt * 4 + hs * 2;            // A: K {0..15|16..31} contiguous per half
    a0.q[0] = Arow0[ai];  a0.q[1] = Arow0[ai + 1];
    a1.q[0] = Arow1[ai];  a1.q[1] = Arow1[ai + 1];
    int q0 = (wn * 32 + l) * 4 + hs * 2;             // 4 uint4 per column
    b0.q[0] = panel[q0];      b0.q[1] = panel[q0 + 1];
    int q1 = (wn * 32 + 16 + l) * 4 + hs * 2;
    b1.q[0] = panel[q1];      b1.q[1] = panel[q1 + 1];

    acc00 = __builtin_amdgcn_wmma_f32_16x16x32_bf16(
        false, a0.v, false, b0.v, (short)0, acc00, false, false);
    acc01 = __builtin_amdgcn_wmma_f32_16x16x32_bf16(
        false, a0.v, false, b1.v, (short)0, acc01, false, false);
    acc10 = __builtin_amdgcn_wmma_f32_16x16x32_bf16(
        false, a1.v, false, b0.v, (short)0, acc10, false, false);
    acc11 = __builtin_amdgcn_wmma_f32_16x16x32_bf16(
        false, a1.v, false, b1.v, (short)0, acc11, false, false);
    __syncthreads();                                 // panel dead -> reusable
  }

  int colBase = bn * 128 + wn * 32;
  int c0 = colBase + l, c1 = colBase + 16 + l;
  float bv0 = bias[c0], bv1 = bias[c1];
  auto emit = [&](float v, size_t oi) {
    if (EPI == 1) {
      float t = tanhf(0.7978845608f * (v + 0.044715f * v * v * v));
      outH[oi] = f2bf(0.5f * v * (1.f + t));
    } else if (EPI == 2) {
      outF[oi] = resid[oi] + v;
    } else {
      outF[oi] = v;
    }
  };
#pragma unroll
  for (int r = 0; r < 8; ++r) {
    int orow0 = bm * 64 + wm * 32 + r + hs * 8;      // C/D: vgpr r -> M=r (+8 hi)
    int orow1 = orow0 + 16;
    emit(acc00[r] + bv0, (size_t)orow0 * N + c0);
    emit(acc01[r] + bv1, (size_t)orow0 * N + c1);
    emit(acc10[r] + bv0, (size_t)orow1 * N + c0);
    emit(acc11[r] + bv1, (size_t)orow1 * N + c1);
  }
}

// --------------------------- ConvSSM: response -----------------------------

__global__ void tanh_prep_kernel(const float* __restrict__ A,
                                 float* __restrict__ out, int n) {
  int id = blockIdx.x * blockDim.x + threadIdx.x;
  if (id < n) out[id] = 0.9f * tanhf(A[id]);
}

// R[f*256+c] = (sum_{k=0..7} A_f^k) * B_f  with wrapped tap positions.
__global__ void ssm_response_kernel(const float* __restrict__ Asta,
                                    const float* __restrict__ Bk,
                                    float* __restrict__ Rre,
                                    float* __restrict__ Rim) {
  int id = blockIdx.x * blockDim.x + threadIdx.x;   // 8192*256
  int c = id & 255;
  int f = id >> 8;
  int fw = f & 31, fh = (f >> 5) & 31, ft = f >> 10;
  const int ttab[3] = {7, 0, 1};                    // (arange(3)-1) % 8
  const int htab[7] = {29, 30, 31, 0, 1, 2, 3};     // (arange(7)-3) % 32
  const float PI2 = 6.28318530718f;
  float etr[3], eti[3], ehr[7], ehi[7], ewr[7], ewi[7];
#pragma unroll
  for (int a = 0; a < 3; ++a) {
    float ang = -PI2 * (float)(ft * ttab[a]) * (1.f / 8.f);
    __sincosf(ang, &eti[a], &etr[a]);
  }
#pragma unroll
  for (int b = 0; b < 7; ++b) {
    float ang = -PI2 * (float)(fh * htab[b]) * (1.f / 32.f);
    __sincosf(ang, &ehi[b], &ehr[b]);
    float ang2 = -PI2 * (float)(fw * htab[b]) * (1.f / 32.f);
    __sincosf(ang2, &ewi[b], &ewr[b]);
  }
  float Ar = 0.f, Aim = 0.f, Br = 0.f, Bi = 0.f;
  const float* Ab = Asta + (size_t)c * 147;
  const float* Bb = Bk + (size_t)c * 147;
  int idx = 0;
  for (int a = 0; a < 3; ++a)
    for (int b = 0; b < 7; ++b) {
      float thr = etr[a] * ehr[b] - eti[a] * ehi[b];
      float thi = etr[a] * ehi[b] + eti[a] * ehr[b];
#pragma unroll
      for (int d = 0; d < 7; ++d) {
        float wr = thr * ewr[d] - thi * ewi[d];
        float wi = thr * ewi[d] + thi * ewr[d];
        float av = Ab[idx], bv = Bb[idx];
        ++idx;
        Ar += av * wr; Aim += av * wi;
        Br += bv * wr; Bi += bv * wi;
      }
    }
  float Sr = 1.f, Si = 0.f;                         // Horner: 1+A(1+A(...))
#pragma unroll
  for (int k = 0; k < 7; ++k) {
    float nr = Sr * Ar - Si * Aim + 1.f;
    Si = Sr * Aim + Si * Ar;
    Sr = nr;
  }
  Rre[id] = Sr * Br - Si * Bi;
  Rim[id] = Sr * Bi + Si * Br;
}

// --------------------------- separable 3D DFT ------------------------------

__global__ void dft_stage_kernel(const float* __restrict__ in_re,
                                 const float* __restrict__ in_im,
                                 float* __restrict__ out_re,
                                 float* __restrict__ out_im,
                                 int outer, int npt, int inner,
                                 float sign, float scale, int realOnly) {
  int id = blockIdx.x * blockDim.x + threadIdx.x;
  int total = outer * npt * inner;
  if (id >= total) return;
  int ii = id % inner;
  int rest = id / inner;
  int fp = rest % npt;
  int oi = rest / npt;
  size_t base = (size_t)oi * npt * inner + ii;
  float ang = sign * 6.28318530718f * (float)fp / (float)npt;
  float sts, stc;
  __sincosf(ang, &sts, &stc);
  float cr = 1.f, ci = 0.f, ar = 0.f, ai = 0.f;
  for (int w = 0; w < npt; ++w) {
    size_t off = base + (size_t)w * inner;
    float vr = in_re[off];
    float vi = in_im ? in_im[off] : 0.f;
    ar += vr * cr - vi * ci;
    ai += vr * ci + vi * cr;
    float nr = cr * stc - ci * sts;                 // rotate twiddle
    ci = cr * sts + ci * stc;
    cr = nr;
  }
  size_t oo = base + (size_t)fp * inner;
  out_re[oo] = ar * scale;
  if (!realOnly) out_im[oo] = ai * scale;
}

__global__ void cmul_kernel(float* __restrict__ xr, float* __restrict__ xi,
                            const float* __restrict__ rr,
                            const float* __restrict__ ri) {
  int id = blockIdx.x * blockDim.x + threadIdx.x;
  float a = xr[id], b = xi[id], c = rr[id], d = ri[id];
  xr[id] = a * c - b * d;
  xi[id] = a * d + b * c;
}

// ------------------------ LayerNorm + residual -----------------------------

__global__ void ln_residual_kernel(const float* __restrict__ Hb,
                                   const float* __restrict__ scale,
                                   const float* __restrict__ bias,
                                   float* __restrict__ X) {
  int pos = blockIdx.x;
  int c = threadIdx.x;
  __shared__ float s1[256];
  __shared__ float s2[256];
  float hv = Hb[(size_t)pos * 256 + c];
  s1[c] = hv;
  s2[c] = hv * hv;
  __syncthreads();
  for (int st = 128; st > 0; st >>= 1) {
    if (c < st) { s1[c] += s1[c + st]; s2[c] += s2[c + st]; }
    __syncthreads();
  }
  float mu  = s1[0] * (1.f / 256.f);
  float var = s2[0] * (1.f / 256.f) - mu * mu;
  float inv = rsqrtf(var + 1e-6f);
  X[(size_t)pos * 256 + c] += (hv - mu) * inv * scale[c] + bias[c];
}

// ------------------------------- heads -------------------------------------

__global__ void mean_kernel(const float* __restrict__ X,
                            float* __restrict__ feat) {
  int t = blockIdx.x, c = threadIdx.x;
  const float* p = X + (size_t)t * 1024 * 256 + c;
  float s = 0.f;
  for (int i = 0; i < 1024; ++i) s += p[(size_t)i * 256];
  feat[t * 256 + c] = s * (1.f / 1024.f);
}

__global__ void head_kernel(const float* __restrict__ feat,
                            const float* __restrict__ qp,
                            const float* __restrict__ tw,
                            const float* __restrict__ tb,
                            const float* __restrict__ ow,
                            const float* __restrict__ ob,
                            float* __restrict__ out) {
  __shared__ float pj[3][8];
  int tid = threadIdx.x;
  if (tid < 24) {
    int t = tid & 7, which = tid >> 3;
    float s = 0.f;
    for (int c = 0; c < 256; ++c) {
      float f = feat[t * 256 + c];
      float w = (which == 0) ? tw[c * 2 + 0]
              : (which == 1) ? tw[c * 2 + 1] : ow[c];
      s += f * w;
    }
    pj[which][t] = s;
  }
  __syncthreads();
  int q = tid;
  float qx = qp[q * 3 + 1], qy = qp[q * 3 + 2];
  for (int t = 0; t < 8; ++t) {
    out[((size_t)q * 8 + t) * 2 + 0] = pj[0][t] + tb[0] + qx;
    out[((size_t)q * 8 + t) * 2 + 1] = pj[1][t] + tb[1] + qy;
    float o = pj[2][t] + ob[0];
    out[4096 + q * 8 + t] = 1.f / (1.f + expf(-o));
  }
}

// ------------------------------- launcher ----------------------------------

extern "C" void kernel_launch(void* const* d_in, const int* in_sizes, int n_in,
                              void* d_out, int out_size, void* d_ws,
                              size_t ws_size, hipStream_t stream) {
  const float* video  = (const float*)d_in[0];
  const float* qp     = (const float*)d_in[1];
  const float* conv_w = (const float*)d_in[2];
  const float* conv_b = (const float*)d_in[3];
  const float* A_k    = (const float*)d_in[4];
  const float* B_k    = (const float*)d_in[5];
  const float* ln_s   = (const float*)d_in[6];
  const float* ln_b   = (const float*)d_in[7];
  const float* w1     = (const float*)d_in[8];
  const float* b1     = (const float*)d_in[9];
  const float* w2     = (const float*)d_in[10];
  const float* b2     = (const float*)d_in[11];
  const float* tw     = (const float*)d_in[12];
  const float* tb     = (const float*)d_in[13];
  const float* ow     = (const float*)d_in[14];
  const float* ob     = (const float*)d_in[15];
  float* out = (float*)d_out;

  char* ws = (char*)d_ws;
  const size_t MB = 1u << 20;
  float* X   = (float*)(ws + 0 * MB);
  float* Hb  = (float*)(ws + 8 * MB);
  float* Are = (float*)(ws + 16 * MB);
  float* Aim = (float*)(ws + 24 * MB);
  float* Bre = (float*)(ws + 32 * MB);
  float* Bim = (float*)(ws + 40 * MB);
  float* Rre = (float*)(ws + 48 * MB);
  float* Rim = (float*)(ws + 56 * MB);
  unsigned short* Mbf = (unsigned short*)(ws + 64 * MB);  // 8192x1024 bf16
  unsigned short* Xbf = (unsigned short*)(ws + 80 * MB);  // 8192x256 bf16
  unsigned short* Wp  = (unsigned short*)(ws + 84 * MB);  // packed weights
  float* Asta = (float*)(ws + 85 * MB);                   // 256x147
  float* feat = (float*)(ws + 86 * MB);                   // 8x256

  const int PC = 8192 * 256;   // pos * channels

  // --- conv (128 -> 256) ---
  f2bf_kernel<<<(8192 * 128 + 255) / 256, 256, 0, stream>>>(video, Xbf,
                                                            8192 * 128);
  pack_w_kernel<<<(4 * 256 * 32 + 255) / 256, 256, 0, stream>>>(conv_w, Wp,
                                                                128, 256);
  gemm_wmma_kernel<0><<<dim3(128, 2), 256, 0, stream>>>(
      Xbf, Wp, conv_b, nullptr, X, nullptr, 8192, 256, 128);

  for (int i = 0; i < 3; ++i) {
    const float* Ai_ = A_k + (size_t)i * 256 * 147;
    const float* Bi_ = B_k + (size_t)i * 256 * 147;
    tanh_prep_kernel<<<(256 * 147 + 255) / 256, 256, 0, stream>>>(
        Ai_, Asta, 256 * 147);
    ssm_response_kernel<<<8192, 256, 0, stream>>>(Asta, Bi_, Rre, Rim);

    // forward 3D DFT of X: W(32) -> H(32) -> T(8)
    dft_stage_kernel<<<8192, 256, 0, stream>>>(X, nullptr, Are, Aim,
                                               256, 32, 256, -1.f, 1.f, 0);
    dft_stage_kernel<<<8192, 256, 0, stream>>>(Are, Aim, Bre, Bim,
                                               8, 32, 8192, -1.f, 1.f, 0);
    dft_stage_kernel<<<8192, 256, 0, stream>>>(Bre, Bim, Are, Aim,
                                               1, 8, 262144, -1.f, 1.f, 0);
    // multiply by response
    cmul_kernel<<<8192, 256, 0, stream>>>(Are, Aim, Rre, Rim);
    // inverse DFT: T -> H -> W (real output)
    dft_stage_kernel<<<8192, 256, 0, stream>>>(Are, Aim, Bre, Bim,
                                               1, 8, 262144, 1.f, 1.f / 8.f, 0);
    dft_stage_kernel<<<8192, 256, 0, stream>>>(Bre, Bim, Are, Aim,
                                               8, 32, 8192, 1.f, 1.f / 32.f, 0);
    dft_stage_kernel<<<8192, 256, 0, stream>>>(Are, Aim, Hb, Bim,
                                               256, 32, 256, 1.f, 1.f / 32.f, 1);
    // x += LN(h)
    ln_residual_kernel<<<8192, 256, 0, stream>>>(Hb, ln_s + i * 256,
                                                 ln_b + i * 256, X);
    // MLP
    f2bf_kernel<<<(PC + 255) / 256, 256, 0, stream>>>(X, Xbf, PC);
    pack_w_kernel<<<(8 * 1024 * 32 + 255) / 256, 256, 0, stream>>>(
        w1 + (size_t)i * 256 * 1024, Wp, 256, 1024);
    gemm_wmma_kernel<1><<<dim3(128, 8), 256, 0, stream>>>(
        Xbf, Wp, b1 + i * 1024, nullptr, nullptr, Mbf, 8192, 1024, 256);
    pack_w_kernel<<<(32 * 256 * 32 + 255) / 256, 256, 0, stream>>>(
        w2 + (size_t)i * 1024 * 256, Wp, 1024, 256);
    gemm_wmma_kernel<2><<<dim3(128, 2), 256, 0, stream>>>(
        Mbf, Wp, b2 + i * 256, X, X, nullptr, 8192, 256, 1024);
  }

  mean_kernel<<<8, 256, 0, stream>>>(X, feat);
  head_kernel<<<1, 256, 0, stream>>>(feat, qp, tw, tb, ow, ob, out);
}
